// QuantScaledDotProductAttention_27101243638406
// MI455X (gfx1250) — 8-bit matrix operand order repaired and checked by reading, compiled, not run on hardware
//
#include <hip/hip_runtime.h>
#include <stdint.h>
#include <math.h>

typedef __attribute__((ext_vector_type(8))) int v8i;
typedef __attribute__((ext_vector_type(4))) int v4i;

#define NH   32
#define SEQ  2048
#define EDIM 128
#define QEPS 1e-8f
#define NEGBIG (-3.0e38f)

__global__ void init_absmax(unsigned* am) {
    if (threadIdx.x < 4) am[threadIdx.x] = 0u;
}

__global__ void absmax_kernel(const float* __restrict__ x, size_t n, unsigned* out) {
    __shared__ unsigned sm[256];
    float m = 0.f;
    for (size_t i = (size_t)blockIdx.x * blockDim.x + threadIdx.x; i < n;
         i += (size_t)gridDim.x * blockDim.x)
        m = fmaxf(m, fabsf(x[i]));
    sm[threadIdx.x] = __float_as_uint(m);
    __syncthreads();
    for (int s = 128; s > 0; s >>= 1) {
        if ((int)threadIdx.x < s) sm[threadIdx.x] = max(sm[threadIdx.x], sm[threadIdx.x + s]);
        __syncthreads();
    }
    if (threadIdx.x == 0) atomicMax(out, sm[0]);
}

__global__ void quant_int8(const float* __restrict__ x, int8_t* __restrict__ y,
                           size_t n, const unsigned* __restrict__ am, float premul) {
    const float s   = fmaxf(__uint_as_float(*am) * premul / 127.f, QEPS);
    const float inv = premul / s;
    for (size_t i = (size_t)blockIdx.x * blockDim.x + threadIdx.x; i < n;
         i += (size_t)gridDim.x * blockDim.x) {
        float r = rintf(x[i] * inv);
        r = fminf(fmaxf(r, -128.f), 127.f);
        y[i] = (int8_t)r;
    }
}

__global__ void quant_vt_int8(const float* __restrict__ x, int8_t* __restrict__ y,
                              size_t n, const unsigned* __restrict__ am) {
    const float s   = fmaxf(__uint_as_float(*am) / 127.f, QEPS);
    const float inv = 1.f / s;
    for (size_t i = (size_t)blockIdx.x * blockDim.x + threadIdx.x; i < n;
         i += (size_t)gridDim.x * blockDim.x) {
        int    e  = (int)(i % EDIM);
        int    sq = (int)((i / EDIM) % SEQ);
        int    nh = (int)(i / ((size_t)EDIM * SEQ));
        float r = rintf(x[i] * inv);
        r = fminf(fmaxf(r, -128.f), 127.f);
        y[((size_t)nh * EDIM + e) * SEQ + sq] = (int8_t)r;
    }
}

union U64x4 { unsigned long long d[4]; v8i v; };
union U32x8 { v4i q[2]; v8i v; };

__device__ __forceinline__ void load_kfrag(const int8_t* krow, int half,
                                           U32x8& kb0, U32x8& kb1) {
    kb0.q[0] = *(const v4i*)(krow +      half * 16);
    kb0.q[1] = *(const v4i*)(krow + 32 + half * 16);
    kb1.q[0] = *(const v4i*)(krow + 64 + half * 16);
    kb1.q[1] = *(const v4i*)(krow + 96 + half * 16);
}

__device__ __forceinline__ void online_update(float f, bool valid, float& m, float& l) {
    float nm = fmaxf(m, f);
    float e  = __expf(fminf(m, f) - nm);
    float lg = l + (valid ? e : 0.f);
    float lr = __builtin_fmaf(l, e, 1.f);
    l = (f > m) ? lr : lg;
    m = nm;
}

__global__ void __launch_bounds__(32)
attn_kernel(const int8_t* __restrict__ Q, const int8_t* __restrict__ K,
            const int8_t* __restrict__ VT, float* __restrict__ O,
            const unsigned* __restrict__ am, float sf) {
    __shared__ __align__(16) uint8_t Plds[16 * 64];

    const int lane = threadIdx.x;
    const int col  = lane & 15;
    const int half = lane >> 4;
    const int m0   = blockIdx.x * 16;
    const int nh   = blockIdx.y;

    const float sq = fmaxf(__uint_as_float(am[0]) * sf / 127.f, QEPS);
    const float sk = fmaxf(__uint_as_float(am[1]) / 127.f, QEPS);
    const float sv = fmaxf(__uint_as_float(am[2]) / 127.f, QEPS);
    const float qkscale = sq * sk;
    const float oscale = sv * (1.f / 255.f);

    const int8_t* qrow = Q + ((size_t)nh * SEQ + m0 + col) * EDIM;
    U64x4 qa[2];
#pragma unroll
    for (int kt = 0; kt < 2; ++kt) {
        const int8_t* p = qrow + kt * 64 + half * 16;
        qa[kt].d[0] = *(const unsigned long long*)(p);
        qa[kt].d[1] = *(const unsigned long long*)(p + 8);
        qa[kt].d[2] = *(const unsigned long long*)(p + 32);
        qa[kt].d[3] = *(const unsigned long long*)(p + 40);
    }

    float mrow[8], lrow[8];
#pragma unroll
    for (int j = 0; j < 8; ++j) { mrow[j] = NEGBIG; lrow[j] = 0.f; }

    const int nblocks = ((m0 + 15) >> 4) + 1;
    const int8_t* kbase = K + (size_t)nh * SEQ * EDIM;
    const v8i z = {};

    int jb = 0;
    for (; jb + 1 < nblocks; jb += 2) {
        const int8_t* krow0 = kbase + (size_t)((jb    ) * 16 + col) * EDIM;
        const int8_t* krow1 = kbase + (size_t)((jb + 1) * 16 + col) * EDIM;
        {
            int jn = (jb + 2 < nblocks) ? jb + 2 : jb;
            __builtin_prefetch(kbase + (size_t)(jn * 16 + col) * EDIM, 0, 3);
        }
        U32x8 a0, a1, b0, b1;
        load_kfrag(krow0, half, a0, a1);
        load_kfrag(krow1, half, b0, b1);
        v8i acc00 = __builtin_amdgcn_wmma_i32_16x16x64_iu8(true, qa[0].v, true, a0.v, z, false, false);
        v8i acc01 = __builtin_amdgcn_wmma_i32_16x16x64_iu8(true, qa[1].v, true, a1.v, z, false, false);
        v8i acc10 = __builtin_amdgcn_wmma_i32_16x16x64_iu8(true, qa[0].v, true, b0.v, z, false, false);
        v8i acc11 = __builtin_amdgcn_wmma_i32_16x16x64_iu8(true, qa[1].v, true, b1.v, z, false, false);

        const int scol0 = jb * 16 + col;
#pragma unroll
        for (int j = 0; j < 8; ++j) {
            const int row = m0 + j + 8 * half;
            bool  v0 = (scol0      <= row);
            float f0 = v0 ? (float)(acc00[j] + acc01[j]) * qkscale : NEGBIG;
            online_update(f0, v0, mrow[j], lrow[j]);
        }
#pragma unroll
        for (int j = 0; j < 8; ++j) {
            const int row = m0 + j + 8 * half;
            bool  v1 = (scol0 + 16 <= row);
            float f1 = v1 ? (float)(acc10[j] + acc11[j]) * qkscale : NEGBIG;
            online_update(f1, v1, mrow[j], lrow[j]);
        }
    }
    if (jb < nblocks) {
        const int8_t* krow = kbase + (size_t)(jb * 16 + col) * EDIM;
        U32x8 a0, a1;
        load_kfrag(krow, half, a0, a1);
        v8i acc0 = __builtin_amdgcn_wmma_i32_16x16x64_iu8(true, qa[0].v, true, a0.v, z, false, false);
        v8i acc1 = __builtin_amdgcn_wmma_i32_16x16x64_iu8(true, qa[1].v, true, a1.v, z, false, false);
        const int scol = jb * 16 + col;
#pragma unroll
        for (int j = 0; j < 8; ++j) {
            const int row = m0 + j + 8 * half;
            bool  v0 = (scol <= row);
            float f  = v0 ? (float)(acc0[j] + acc1[j]) * qkscale : NEGBIG;
            online_update(f, v0, mrow[j], lrow[j]);
        }
    }

#pragma unroll
    for (int j = 0; j < 8; ++j) {
        float m = mrow[j], l = lrow[j];
#pragma unroll
        for (int msk = 1; msk <= 8; msk <<= 1) {
            float mo = __shfl_xor(m, msk, 16);
            float lo = __shfl_xor(l, msk, 16);
            float nm = fmaxf(m, mo);
            l = l * __expf(m - nm) + lo * __expf(mo - nm);
            m = nm;
        }
        mrow[j] = m; lrow[j] = l;
    }

    float linv[8];
#pragma unroll
    for (int j = 0; j < 8; ++j) linv[j] = 255.f / lrow[j];

    v8i acco[8];
#pragma unroll
    for (int t = 0; t < 8; ++t) acco[t] = z;

    const int sbmax = (m0 + 15) >> 6;
    for (int sb = 0; sb <= sbmax; ++sb) {
        const int sbase = sb * 64;
#pragma unroll
        for (int jj = 0; jj < 4; ++jj) {
            const int8_t* krow = kbase + (size_t)(sbase + jj * 16 + col) * EDIM;
            U32x8 kb0, kb1;
            load_kfrag(krow, half, kb0, kb1);
            v8i acc0 = __builtin_amdgcn_wmma_i32_16x16x64_iu8(true, qa[0].v, true, kb0.v, z, false, false);
            v8i acc1 = __builtin_amdgcn_wmma_i32_16x16x64_iu8(true, qa[1].v, true, kb1.v, z, false, false);

            const int scol = sbase + jj * 16 + col;
#pragma unroll
            for (int j = 0; j < 8; ++j) {
                const int row = m0 + j + 8 * half;
                float f = (scol <= row) ? (float)(acc0[j] + acc1[j]) * qkscale : NEGBIG;
                float p = __expf(f - mrow[j]) * linv[j];
                p = fminf(fmaxf(rintf(p), 0.f), 255.f);
                Plds[(j + 8 * half) * 64 + jj * 16 + col] = (uint8_t)p;
            }
        }
        __syncthreads();

        U64x4 pa;
        {
            const uint8_t* pr = &Plds[col * 64 + half * 16];
            pa.d[0] = *(const unsigned long long*)(pr);
            pa.d[1] = *(const unsigned long long*)(pr + 8);
            pa.d[2] = *(const unsigned long long*)(pr + 32);
            pa.d[3] = *(const unsigned long long*)(pr + 40);
        }
#pragma unroll
        for (int t = 0; t < 8; ++t) {
            const int8_t* vrow = VT + ((size_t)nh * EDIM + t * 16 + col) * SEQ;
            U32x8 vb;
            vb.q[0] = *(const v4i*)(vrow + sbase + half * 16);
            vb.q[1] = *(const v4i*)(vrow + sbase + 32 + half * 16);
            acco[t] = __builtin_amdgcn_wmma_i32_16x16x64_iu8(
                          false  , pa.v, true  , vb.v,
                          acco[t], false, false);
        }
        __syncthreads();
    }

#pragma unroll
    for (int t = 0; t < 8; ++t)
#pragma unroll
        for (int j = 0; j < 8; ++j)
            O[((size_t)nh * SEQ + m0 + j + 8 * half) * EDIM + t * 16 + col] =
                (float)acco[t][j] * oscale;
}

extern "C" void kernel_launch(void* const* d_in, const int* in_sizes, int n_in,
                              void* d_out, int out_size, void* d_ws, size_t ws_size,
                              hipStream_t stream) {
    (void)in_sizes; (void)n_in; (void)out_size; (void)ws_size;
    const float* q = (const float*)d_in[0];
    const float* k = (const float*)d_in[1];
    const float* v = (const float*)d_in[2];
    float* out = (float*)d_out;

    const size_t n = (size_t)NH * SEQ * EDIM;

    uint8_t* ws   = (uint8_t*)d_ws;
    unsigned* am  = (unsigned*)ws;
    int8_t*  Qi8  = (int8_t*)(ws + 16);
    int8_t*  Ki8  = Qi8 + n;
    int8_t*  Vt   = Ki8 + n;

    const float sf = 1.0f / sqrtf((float)EDIM);

    init_absmax<<<1, 32, 0, stream>>>(am);
    absmax_kernel<<<1024, 256, 0, stream>>>(q, n, am + 0);
    absmax_kernel<<<1024, 256, 0, stream>>>(k, n, am + 1);
    absmax_kernel<<<1024, 256, 0, stream>>>(v, n, am + 2);

    quant_int8   <<<4096, 256, 0, stream>>>(q, Qi8, n, am + 0, sf);
    quant_int8   <<<4096, 256, 0, stream>>>(k, Ki8, n, am + 1, 1.0f);
    quant_vt_int8<<<4096, 256, 0, stream>>>(v, Vt, n, am + 2);

    dim3 grid(SEQ / 16, NH);
    attn_kernel<<<grid, 32, 0, stream>>>(Qi8, Ki8, Vt, out, am, sf);
}
